// MSTPFormerBlock_18476949307579
// MI455X (gfx1250) — compile-verified, matrix operand order checked by reading, not run on hardware
//
#include <hip/hip_runtime.h>
#include <hip/hip_bf16.h>
#include <math.h>

// ---------------------------------------------------------------------------
// MSTPFormer block for MI455X (gfx1250, wave32, WMMA + async global->LDS).
// Memory-bound (~675 MB HBM traffic ~ 29us @ 23.3 TB/s); GEMM done with
// v_wmma_f32_16x16x32_bf16; x slab staged to LDS via global_load_async_to_lds
// (ASYNCcnt) so norms/gate/Gram use exact f32 while the DMA overlaps W setup.
// ---------------------------------------------------------------------------

#define JN 17
#define CD 128
#define CP 136          // bf16 LDS row stride: 272B -> bank rotate
#define CF 132          // f32 staging row stride: 528B -> bank rotate
#define GS 16           // samples per block
#define ROWS (GS*JN)    // 272 rows = 17 exact 16-row WMMA M-tiles
#define NTOT 15552      // B*T
#define NBLK (NTOT/GS)  // 972
#define PAIRS 153       // 17*18/2
#define NT_ELEMS (NTOT*JN)

// workspace layout (float units)
#define WS_COMB 0       // 289 (padded to 320)
#define WS_SC   320     // 128: invstd*gamma
#define WS_SH   448     // 128: beta - mean*scale
#define WS_PART 576     // mode0: NBLK*256 partials | mode1: 256 atomic accum

// LDS layout (bytes)
#define L_X    0                         // 272*136*2 = 73984  (bf16 x)
#define L_W    73984                     // 128*136*2 = 34816  (bf16 W)
#define L_XF   108800                    // 272*132*4 = 143616 (f32 x staging,
                                         //  later reused as 272*128*4 f32 x@W)
#define L_A    252416                    // 16*17*17*4 = 18496 (A_hat)
#define L_RN   270912                    // 272*4
#define L_GT   272000                    // 272*4
#define L_DG   273088                    // 272*4
#define L_CB   274176                    // 292*4 (comb, padded)
#define L_GW   275344                    // 128*4 (gate_w)
#define L_B    275856                    // 128*4 (bias)
#define L_RED  276368                    // 8*32*8*4 = 8192 (BN reduce)
#define SMEM_BYTES 284560

typedef __attribute__((ext_vector_type(16))) __bf16 v16bf;
typedef __attribute__((ext_vector_type(8)))  float  v8f;

union FragBF {                    // 32B: one WMMA bf16 operand (8 VGPRs)
  uint4          u[2];
  unsigned short s[16];
  v16bf          v;
};

__device__ __forceinline__ unsigned short f2bf(float f) {  // round-to-nearest-even
  unsigned int u; __builtin_memcpy(&u, &f, 4);
  return (unsigned short)((u + 0x7fffu + ((u >> 16) & 1u)) >> 16);
}
__device__ __forceinline__ float sigmoidf_(float x) { return 1.0f / (1.0f + expf(-x)); }
__device__ __forceinline__ float softplusf_(float x) {
  return (x > 20.0f) ? x : log1pf(expf(x));
}

// CDNA5 async global->LDS copy, 16B per lane, tracked by ASYNCcnt.
// lds_addr = byte offset within workgroup LDS; goff = byte offset vs saddr.
__device__ __forceinline__ void async_g2lds_b128(unsigned lds_addr, unsigned goff,
                                                 const void* saddr) {
  asm volatile("global_load_async_to_lds_b128 %0, %1, %2"
               :: "v"(lds_addr), "v"(goff), "s"(saddr) : "memory");
}
__device__ __forceinline__ void wait_async0() {
  asm volatile("s_wait_asynccnt 0" ::: "memory");
}

// ---------------------------------------------------------------------------
// Kernel 0: symmetrized learnable static adjacency  comb = 0.5*(M + M^T)
// ---------------------------------------------------------------------------
__global__ void k_comb(const float* __restrict__ l1, const float* __restrict__ l2,
                       const float* __restrict__ w1, const float* __restrict__ w2,
                       const float* __restrict__ a1, const float* __restrict__ a2,
                       float* __restrict__ ws) {
  float sp1 = softplusf_(w1[0]), sp2 = softplusf_(w2[0]);
  for (int p = threadIdx.x; p < JN * JN; p += blockDim.x) {
    int i = p / JN, j = p % JN, q = j * JN + i;
    float f = sp1 * (a1[p] + sigmoidf_(l1[p])) + sp2 * (a2[p] + sigmoidf_(l2[p]));
    float g = sp1 * (a1[q] + sigmoidf_(l1[q])) + sp2 * (a2[q] + sigmoidf_(l2[q]));
    ws[WS_COMB + p] = 0.5f * (f + g);
  }
}

// ---------------------------------------------------------------------------
// Kernel 1: per-block (16 samples): adjacency + WMMA GEMM + A_hat apply
// ---------------------------------------------------------------------------
__global__ void __launch_bounds__(256)
k_main(const float* __restrict__ x, const float* __restrict__ W,
       const float* __restrict__ bias, const float* __restrict__ gate_w,
       const float* __restrict__ gate_b, float* __restrict__ out,
       float* __restrict__ ws, int mode) {
  extern __shared__ char smem[];
  unsigned short* sXu = (unsigned short*)(smem + L_X);
  unsigned short* sWu = (unsigned short*)(smem + L_W);
  float* sXf = (float*)(smem + L_XF);   // f32 x staging (async DMA target)
  float* sXW = (float*)(smem + L_XF);   // later: f32 x@W (same region)
  float* sA  = (float*)(smem + L_A);
  float* sRn = (float*)(smem + L_RN);
  float* sGt = (float*)(smem + L_GT);
  float* sDg = (float*)(smem + L_DG);
  float* sCb = (float*)(smem + L_CB);
  float* sGw = (float*)(smem + L_GW);
  float* sB  = (float*)(smem + L_B);
  float* sRed = (float*)(smem + L_RED);

  const int t = threadIdx.x;
  const int n0 = blockIdx.x * GS;

  // ---- phase 0a: kick off async DMA of the x slab into f32 LDS staging ----
  {
    const float* xbase = x + (size_t)n0 * JN * CD;
    #pragma unroll
    for (int k = 0; k < 34; ++k) {                 // 272*128/4 = 8704 float4
      int idx = t + k * 256;
      int row = idx >> 5, col4 = (idx & 31) << 2;
      unsigned lds_addr = (unsigned)(L_XF + (row * CF + col4) * 4);
      async_g2lds_b128(lds_addr, (unsigned)idx * 16u, xbase);
    }
    // overlap: W f32 -> bf16 LDS while the DMA streams
    const float4* wg = (const float4*)W;
    #pragma unroll
    for (int k = 0; k < 16; ++k) {                 // 128*128/4 = 4096 float4
      int idx = t + k * 256;
      float4 v = wg[idx];
      int row = idx >> 5, col4 = (idx & 31) << 2;
      ushort4 h = make_ushort4(f2bf(v.x), f2bf(v.y), f2bf(v.z), f2bf(v.w));
      *(ushort4*)(sWu + row * CP + col4) = h;
    }
    for (int p = t; p < JN * JN; p += 256) sCb[p] = ws[WS_COMB + p];
    if (t < CD) { sGw[t] = gate_w[t]; sB[t] = bias[t]; }
    wait_async0();
  }
  __syncthreads();

  // ---- phase 0b: convert staged f32 x -> padded bf16 for WMMA A ----
  #pragma unroll
  for (int k = 0; k < 34; ++k) {
    int idx = t + k * 256;
    int row = idx >> 5, col4 = (idx & 31) << 2;
    float4 v = *(const float4*)(sXf + row * CF + col4);
    ushort4 h = make_ushort4(f2bf(v.x), f2bf(v.y), f2bf(v.z), f2bf(v.w));
    *(ushort4*)(sXu + row * CP + col4) = h;
  }

  const float gb = gate_b[0];

  // ---- phase 1: row norms and gate values (exact f32) ----
  for (int r = t; r < ROWS; r += 256) {
    const float4* xr = (const float4*)(sXf + r * CF);
    float ss = 0.f, gd = 0.f;
    #pragma unroll 8
    for (int k = 0; k < 32; ++k) {
      float4 q = xr[k];
      const float* g = sGw + k * 4;
      ss += q.x*q.x + q.y*q.y + q.z*q.z + q.w*q.w;
      gd += q.x*g[0] + q.y*g[1] + q.z*g[2] + q.w*g[3];
    }
    sRn[r] = 1.0f / fmaxf(sqrtf(ss), 1e-12f);
    sGt[r] = sigmoidf_(gd + gb);
  }
  __syncthreads();

  // ---- phase 2a: symmetric cosine Gram (dyn adjacency), f32 exact ----
  for (int p = t; p < GS * PAIRS; p += 256) {
    int s = p / PAIRS, pp = p - s * PAIRS;
    int i = 0, rem = pp;
    while (rem >= JN - i) { rem -= JN - i; ++i; }
    int j = i + rem;
    const float4* ui = (const float4*)(sXf + (s * JN + i) * CF);
    const float4* uj = (const float4*)(sXf + (s * JN + j) * CF);
    float d = 0.f;
    #pragma unroll 8
    for (int k = 0; k < 32; ++k) {
      float4 a = ui[k], b = uj[k];
      d += a.x*b.x + a.y*b.y + a.z*b.z + a.w*b.w;
    }
    float cs = d * sRn[s * JN + i] * sRn[s * JN + j];
    float dyn = fmaxf(cs, 0.f) + ((i == j) ? 1.0f : 0.0f);
    sA[(s * JN + i) * JN + j] = dyn;
    sA[(s * JN + j) * JN + i] = dyn;
  }
  __syncthreads();

  // ---- phase 2b: gate-mix with static comb, degree ----
  for (int r = t; r < ROWS; r += 256) {
    int i = r % JN;
    float g = sGt[r], deg = 0.f;
    float* Ar = sA + r * JN;
    const float* cb = sCb + i * JN;
    #pragma unroll
    for (int j = 0; j < JN; ++j) {
      float a = g * cb[j] + (1.f - g) * Ar[j];
      Ar[j] = a; deg += a;
    }
    sDg[r] = 1.0f / sqrtf(deg + 1e-6f);
  }
  __syncthreads();

  // ---- phase 2c: symmetric normalization A_hat = d_i * A * d_j ----
  for (int r = t; r < ROWS; r += 256) {
    int s = r / JN;
    float di = sDg[r];
    float* Ar = sA + r * JN;
    const float* dj = sDg + s * JN;
    #pragma unroll
    for (int j = 0; j < JN; ++j) Ar[j] *= di * dj[j];
  }
  __syncthreads();

  // ---- phase 3: xW = x @ W via v_wmma_f32_16x16x32_bf16 ----
  // wave w owns N-tile w (cols 16w..16w+15); B fragments cached in VGPRs.
  // NOTE: overwrites the f32 staging region (all f32-x reads are done).
  {
    const int lane = t & 31, wave = t >> 5;
    const int half = lane >> 4, mrow = lane & 15;
    const int c0 = wave * 16;
    FragBF Bf[4];
    #pragma unroll
    for (int kb = 0; kb < 4; ++kb) {              // K base = 32*kb
      int kbase = kb * 32 + 8 * half;             // ISA 7.12.2 bf16 layout
      #pragma unroll
      for (int e = 0; e < 8; ++e) Bf[kb].s[e]     = sWu[(kbase + e) * CP + c0 + mrow];
      #pragma unroll
      for (int e = 0; e < 8; ++e) Bf[kb].s[8 + e] = sWu[(kbase + 16 + e) * CP + c0 + mrow];
    }
    for (int mt = 0; mt < 17; ++mt) {             // 17 M-tiles of 16 rows
      const int row = mt * 16 + mrow;
      v8f acc = {0.f, 0.f, 0.f, 0.f, 0.f, 0.f, 0.f, 0.f};
      #pragma unroll
      for (int kb = 0; kb < 4; ++kb) {
        FragBF Af;
        const uint4* src = (const uint4*)(sXu + row * CP + kb * 32 + 8 * half);
        Af.u[0] = src[0];                         // K: kb*32+8h .. +7
        Af.u[1] = src[2];                         // K: kb*32+16+8h .. +7
        acc = __builtin_amdgcn_wmma_f32_16x16x32_bf16(
            false, Af.v, false, Bf[kb].v, (short)0, acc, false, false);
      }
      #pragma unroll
      for (int r8 = 0; r8 < 8; ++r8)              // C/D layout: lane= N, vgpr= M
        sXW[(mt * 16 + r8 + 8 * half) * CD + c0 + mrow] = acc[r8];
    }
  }
  __syncthreads();

  // ---- phase 4: h = A_hat @ xW + b -> d_out, + deterministic BN partials ----
  {
    const int c4 = t & 31, rg = t >> 5;           // fixed channel group per thread
    float4 b4 = ((const float4*)sB)[c4];
    float sx0=0,sx1=0,sx2=0,sx3=0, q0=0,q1=0,q2=0,q3=0;
    for (int k = 0; k < 34; ++k) {
      int r = rg + 8 * k;
      int s = r / JN, i = r - s * JN;
      const float* Ar = sA + r * JN;
      const float* xwb = sXW + (s * JN) * CD + c4 * 4;
      float4 acc = b4;
      #pragma unroll
      for (int j = 0; j < JN; ++j) {
        float a = Ar[j];
        float4 w4 = *(const float4*)(xwb + j * CD);
        acc.x += a * w4.x; acc.y += a * w4.y; acc.z += a * w4.z; acc.w += a * w4.w;
      }
      ((float4*)out)[((size_t)(n0 + s) * JN + i) * 32 + c4] = acc;
      sx0 += acc.x; sx1 += acc.y; sx2 += acc.z; sx3 += acc.w;
      q0 += acc.x*acc.x; q1 += acc.y*acc.y; q2 += acc.z*acc.z; q3 += acc.w*acc.w;
    }
    float4* red = (float4*)sRed;
    red[(rg * 32 + c4) * 2 + 0] = make_float4(sx0, sx1, sx2, sx3);
    red[(rg * 32 + c4) * 2 + 1] = make_float4(q0, q1, q2, q3);
  }
  __syncthreads();
  if (t < 32) {
    float4* red = (float4*)sRed;
    float4 s = make_float4(0,0,0,0), q = make_float4(0,0,0,0);
    #pragma unroll
    for (int rg = 0; rg < 8; ++rg) {
      float4 a = red[(rg * 32 + t) * 2 + 0], b = red[(rg * 32 + t) * 2 + 1];
      s.x += a.x; s.y += a.y; s.z += a.z; s.w += a.w;
      q.x += b.x; q.y += b.y; q.z += b.z; q.w += b.w;
    }
    if (mode == 0) {
      float* dst = ws + WS_PART + (size_t)blockIdx.x * 256;
      dst[4*t+0]=s.x; dst[4*t+1]=s.y; dst[4*t+2]=s.z; dst[4*t+3]=s.w;
      dst[128+4*t+0]=q.x; dst[128+4*t+1]=q.y; dst[128+4*t+2]=q.z; dst[128+4*t+3]=q.w;
    } else {
      float* dst = ws + WS_PART;
      atomicAdd(&dst[4*t+0], s.x); atomicAdd(&dst[4*t+1], s.y);
      atomicAdd(&dst[4*t+2], s.z); atomicAdd(&dst[4*t+3], s.w);
      atomicAdd(&dst[128+4*t+0], q.x); atomicAdd(&dst[128+4*t+1], q.y);
      atomicAdd(&dst[128+4*t+2], q.z); atomicAdd(&dst[128+4*t+3], q.w);
    }
  }
}

// ---------------------------------------------------------------------------
// Kernel 2: fold partials -> per-channel scale/shift for BN
// ---------------------------------------------------------------------------
__global__ void k_stats(float* __restrict__ ws, const float* __restrict__ gamma,
                        const float* __restrict__ beta, int mode) {
  int c = threadIdx.x;                            // 128 threads
  float sum = 0.f, ssq = 0.f;
  if (mode == 0) {
    for (int b = 0; b < NBLK; ++b) {
      const float* p = ws + WS_PART + (size_t)b * 256;
      sum += p[c]; ssq += p[128 + c];
    }
  } else {
    sum = ws[WS_PART + c]; ssq = ws[WS_PART + 128 + c];
  }
  float inv = 1.0f / (float)NT_ELEMS;
  float mean = sum * inv;
  float var = fmaxf(ssq * inv - mean * mean, 0.f);
  float sc = gamma[c] / sqrtf(var + 1e-5f);
  ws[WS_SC + c] = sc;
  ws[WS_SH + c] = beta[c] - mean * sc;
}

// ---------------------------------------------------------------------------
// Kernel 3: out = relu(h*sc + sh) + x   (h in d_out is L2-resident, 192MB L2)
// ---------------------------------------------------------------------------
__global__ void __launch_bounds__(256)
k_final(float* __restrict__ out, const float* __restrict__ x,
        const float* __restrict__ ws) {
  size_t e = (size_t)blockIdx.x * 256 + threadIdx.x;   // float4 index
  int c4 = (int)(e & 31);
  float4 h = ((const float4*)out)[e];
  float4 xv = ((const float4*)x)[e];
  float4 sc = ((const float4*)(ws + WS_SC))[c4];
  float4 sh = ((const float4*)(ws + WS_SH))[c4];
  float4 o;
  o.x = fmaxf(h.x * sc.x + sh.x, 0.f) + xv.x;
  o.y = fmaxf(h.y * sc.y + sh.y, 0.f) + xv.y;
  o.z = fmaxf(h.z * sc.z + sh.z, 0.f) + xv.z;
  o.w = fmaxf(h.w * sc.w + sh.w, 0.f) + xv.w;
  ((float4*)out)[e] = o;
}

// ---------------------------------------------------------------------------
extern "C" void kernel_launch(void* const* d_in, const int* in_sizes, int n_in,
                              void* d_out, int out_size, void* d_ws, size_t ws_size,
                              hipStream_t stream) {
  (void)in_sizes; (void)n_in; (void)out_size;
  const float* x     = (const float*)d_in[0];
  const float* W     = (const float*)d_in[1];
  const float* b     = (const float*)d_in[2];
  const float* l1    = (const float*)d_in[3];
  const float* l2    = (const float*)d_in[4];
  const float* w1    = (const float*)d_in[5];
  const float* w2    = (const float*)d_in[6];
  const float* gw    = (const float*)d_in[7];
  const float* gb    = (const float*)d_in[8];
  const float* gamma = (const float*)d_in[9];
  const float* beta  = (const float*)d_in[10];
  const float* a1    = (const float*)d_in[11];
  const float* a2    = (const float*)d_in[12];
  float* out = (float*)d_out;
  float* ws  = (float*)d_ws;

  // deterministic per-block partials if ws is big enough; atomic fallback else
  size_t need = (size_t)(WS_PART + NBLK * 256) * sizeof(float);
  int mode = (ws_size >= need) ? 0 : 1;
  if (mode == 1)
    hipMemsetAsync((char*)d_ws + WS_PART * sizeof(float), 0, 256 * sizeof(float), stream);

  k_comb<<<1, 64, 0, stream>>>(l1, l2, w1, w2, a1, a2, ws);
  k_main<<<NBLK, 256, SMEM_BYTES, stream>>>(x, W, b, gw, gb, out, ws, mode);
  k_stats<<<1, 128, 0, stream>>>(ws, gamma, beta, mode);
  const int n4 = (NTOT * JN * CD) / 4;            // 8,460,288 float4
  k_final<<<n4 / 256, 256, 0, stream>>>(out, x, ws);
}
